// DescrptSeA_55482387529786
// MI455X (gfx1250) — matrix operand order repaired and checked by reading, not compiled, not run on hardware
//
#include <hip/hip_runtime.h>
#include <hip/hip_bf16.h>

typedef __attribute__((ext_vector_type(16))) _Float16 v16h;
typedef __attribute__((ext_vector_type(8)))  _Float16 v8h;
typedef __attribute__((ext_vector_type(2)))  _Float16 h2v;
typedef __attribute__((ext_vector_type(8)))  float    v8f;

#define NF    4
#define NLOC  4096
#define NALL  4096
#define NNEI  128
#define AXIS  8
#define NSITES (NF * NLOC)

// padded row strides (halves): +8 keeps 16B row alignment, staggers banks
#define H1S  40   // sH1 [128][40], K used 0..24 (24 = bias-one slot), 25..31 zero
#define H2S  72   // sH2 [128][72], K used 0..48 (48 = bias-one slot), 49..63 zero
#define H2TS 136  // sH2t [48][136]
#define GTS  136  // sGt  [96][136]
#define ETS  136  // sEnvT[16][136]
#define W2S  40   // sW2t [48][40]
#define W3S  72   // sW3t [96][72]

// Hardware v_tanh_f32 (confirmed in codegen); branch-free fallback otherwise.
__device__ __forceinline__ float fast_tanh(float x) {
#if __has_builtin(__builtin_amdgcn_tanhf)
  return __builtin_amdgcn_tanhf(x);
#else
  const float ax = __builtin_fabsf(x);
  const float t  = __builtin_amdgcn_exp2f(-2.885390081777927f * ax);
  const float r  = __builtin_amdgcn_rcpf(1.0f + t);
  return __builtin_copysignf((1.0f - t) * r, x);
#endif
}

// A-operand: 16x32 f16 tile (MxK), contiguous k = 16*(lane/16) + s (pairs -> b128 merge).
__device__ __forceinline__ v16h load_a_tile(const _Float16* base, int strideH,
                                            int m0, int kbase, int lane) {
  const int lrow = lane & 15;
  const int khi  = (lane & 16) ? 16 : 0;
  const _Float16* row = base + (m0 + lrow) * strideH + kbase;
  v16h a;
#pragma unroll
  for (int v = 0; v < 8; ++v) {
    const int kb = 2 * v + khi;
    a[2 * v]     = row[kb];
    a[2 * v + 1] = row[kb + 1];
  }
  return a;
}

// B-operand: 32x16 f16 tile (KxN) from LDS stored [N][K].
__device__ __forceinline__ v16h load_b_tile(const _Float16* base, int strideH,
                                            int n0, int kbase, int lane) {
  const int nn = lane & 15;
  const int kg = (lane & 16) ? 16 : 0;
  const _Float16* row = base + (n0 + nn) * strideH + kbase + kg;
  v16h b;
#pragma unroll
  for (int v = 0; v < 8; ++v) {
    b[2 * v]     = row[2 * v];
    b[2 * v + 1] = row[2 * v + 1];
  }
  return b;
}

__device__ __forceinline__ v8f wmma_f16(v16h a, v16h b, v8f c) {
  return __builtin_amdgcn_wmma_f32_16x16x32_f16(false, a, false, b, (short)0, c,
                                                false, false);
}

__global__ __launch_bounds__(128) void descrpt_se_a_kernel(
    const float* __restrict__ coord, const float* __restrict__ davg,
    const float* __restrict__ dstd,  const float* __restrict__ w1,
    const float* __restrict__ b1,    const float* __restrict__ w2,
    const float* __restrict__ b2,    const float* __restrict__ w3,
    const float* __restrict__ b3,    const int* __restrict__ nlist,
    float* __restrict__ out) {
  __shared__ _Float16 sW2t[48][W2S];    // layer2 B [N][K]: K24=b2, 25..31=0
  __shared__ _Float16 sW3t[96][W3S];    // layer3 B [N][K]: K48=b3, 49..63=0
  __shared__ _Float16 sH1[128][H1S];    // layer2 A: K24=1.0 (bias row), 25..31=0
  __shared__ _Float16 sH2[128][H2S];    // layer3 A: K48=1.0 (bias row), 49..63=0
  __shared__ _Float16 sH2t[48][H2TS];   // h2 transposed: b128 skip reads in L3
  __shared__ _Float16 sGt[96][GTS];     // G transposed: B operand for T
  __shared__ _Float16 sEnvT[16][ETS];   // env^T (M=4 padded to 16): A for T
  __shared__ float    sTpart[4][4][96];
  __shared__ float    sT[4][96];
  __shared__ float    sW1[24], sB1[24];
  __shared__ float    sDavg[128][4];
  __shared__ float    sRstd[128][4];

  const int tid  = threadIdx.x;   // 0..127, == neighbor slot in phase A
  const int lane = tid & 31;
  const int wave = tid >> 5;      // wave w owns neighbors 32w..32w+31

  // ---- one-time setup ----
  for (int idx = tid; idx < 48 * 32; idx += 128) {
    const int nn = idx >> 5, k = idx & 31;
    float v = 0.0f;
    if (k < 24) v = w2[k * 48 + nn];
    else if (k == 24) v = b2[nn];           // bias folded into pad row
    sW2t[nn][k] = (_Float16)v;
  }
  for (int idx = tid; idx < 96 * 64; idx += 128) {
    const int nn = idx >> 6, k = idx & 63;
    float v = 0.0f;
    if (k < 48) v = w3[k * 96 + nn];
    else if (k == 48) v = b3[nn];           // bias folded into pad row
    sW3t[nn][k] = (_Float16)v;
  }
  if (tid < 24) { sW1[tid] = w1[tid]; sB1[tid] = b1[tid]; }
#pragma unroll
  for (int c = 0; c < 4; ++c) {
    sDavg[tid][c] = davg[tid * 4 + c];
    sRstd[tid][c] = 1.0f / dstd[tid * 4 + c];
  }
  // A pad slots: ones column feeds the bias row of B; rest of read range zero
  sH1[tid][24] = (_Float16)1.0f;
#pragma unroll
  for (int k = 25; k < 32; ++k) sH1[tid][k] = (_Float16)0.0f;
  sH2[tid][48] = (_Float16)1.0f;
#pragma unroll
  for (int k = 49; k < 64; ++k) sH2[tid][k] = (_Float16)0.0f;
  for (int idx = tid; idx < 12 * 128; idx += 128)
    sEnvT[4 + (idx >> 7)][idx & 127] = (_Float16)0.0f;
  __syncthreads();

  // persistent B-operand register tiles
  v16h B2[3], B3[6][2];
#pragma unroll
  for (int nt = 0; nt < 3; ++nt)
    B2[nt] = load_b_tile(&sW2t[0][0], W2S, nt * 16, 0, lane);
#pragma unroll
  for (int nt = 0; nt < 6; ++nt) {
    B3[nt][0] = load_b_tile(&sW3t[0][0], W3S, nt * 16, 0, lane);
    B3[nt][1] = load_b_tile(&sW3t[0][0], W3S, nt * 16, 32, lane);
  }

  for (int site = blockIdx.x; site < NSITES; site += gridDim.x) {
    const int f = site >> 12;
    const int i = site & (NLOC - 1);

    // ---- phase A: env row + layer1 (one neighbor per thread) ----
    {
      const int n = tid;
      const int nsite = site + (int)gridDim.x;
      if (nsite < NSITES)
        __builtin_prefetch(&nlist[(size_t)nsite * NNEI + n], 0, 1);
      const int j = nlist[(size_t)site * NNEI + n];
      const float px = coord[(f * NALL + i) * 3 + 0];
      const float py = coord[(f * NALL + i) * 3 + 1];
      const float pz = coord[(f * NALL + i) * 3 + 2];
      const float rx = coord[(f * NALL + j) * 3 + 0] - px;
      const float ry = coord[(f * NALL + j) * 3 + 1] - py;
      const float rz = coord[(f * NALL + j) * 3 + 2] - pz;
      const float r     = sqrtf(rx * rx + ry * ry + rz * rz + 1e-12f);
      const float inv_r = 1.0f / r;
      const float uu    = (r - 0.5f) * (1.0f / 5.5f);
      const float swm   = uu * uu * uu * (-6.0f * uu * uu + 15.0f * uu - 10.0f) + 1.0f;
      const float sw    = (r < 0.5f) ? 1.0f : ((r < 6.0f) ? swm : 0.0f);
      const float s     = sw * inv_r;
      const float e0 = (s              - sDavg[n][0]) * sRstd[n][0];
      const float e1 = (s * rx * inv_r - sDavg[n][1]) * sRstd[n][1];
      const float e2 = (s * ry * inv_r - sDavg[n][2]) * sRstd[n][2];
      const float e3 = (s * rz * inv_r - sDavg[n][3]) * sRstd[n][3];
      sEnvT[0][n] = (_Float16)e0;
      sEnvT[1][n] = (_Float16)e1;
      sEnvT[2][n] = (_Float16)e2;
      sEnvT[3][n] = (_Float16)e3;
#pragma unroll
      for (int k = 0; k < 24; k += 2) {
        h2v p;
        p[0] = (_Float16)fast_tanh(e0 * sW1[k]     + sB1[k]);
        p[1] = (_Float16)fast_tanh(e0 * sW1[k + 1] + sB1[k + 1]);
        *(h2v*)&sH1[n][k] = p;
      }
    }

    // ---- layer 2: bias via pad row; batch 3 WMMAs, then epilogues ----
#pragma unroll
    for (int mt = 0; mt < 2; ++mt) {
      const int m0 = (wave * 2 + mt) * 16;
      const v16h A1 = load_a_tile(&sH1[0][0], H1S, m0, 0, lane);
      v8f c[3];
#pragma unroll
      for (int nt = 0; nt < 3; ++nt) { v8f z = {}; c[nt] = wmma_f16(A1, B2[nt], z); }
      const int mrow = m0 + ((lane >> 4) << 3);
#pragma unroll
      for (int nt = 0; nt < 3; ++nt) {
        const int colg = nt * 16 + (lane & 15);
        const int scol = (colg < 24) ? colg : colg - 24;
        v8h pack;                            // single conversion per element
#pragma unroll
        for (int r = 0; r < 8; ++r)
          pack[r] = (_Float16)(fast_tanh(c[nt][r]) + (float)sH1[mrow + r][scol]);
        *(v8h*)&sH2t[colg][mrow] = pack;     // transposed: layer3 skip reads
#pragma unroll
        for (int r = 0; r < 8; ++r)          // row-major from pack halves
          sH2[mrow + r][colg] = pack[r];     // (b16 / b16_d16_hi extracts)
      }
    }

    // ---- layer 3: batch 3 tiles (2 WMMA each), then epilogues ----
#pragma unroll
    for (int mt = 0; mt < 2; ++mt) {
      const int m0 = (wave * 2 + mt) * 16;
      const v16h A2a = load_a_tile(&sH2[0][0], H2S, m0, 0, lane);
      const v16h A2b = load_a_tile(&sH2[0][0], H2S, m0, 32, lane);
      const int mrow = m0 + ((lane >> 4) << 3);
#pragma unroll
      for (int g = 0; g < 2; ++g) {
        v8f c[3];
#pragma unroll
        for (int t = 0; t < 3; ++t) {
          v8f z = {};
          z = wmma_f16(A2a, B3[g * 3 + t][0], z);
          c[t] = wmma_f16(A2b, B3[g * 3 + t][1], z);
        }
#pragma unroll
        for (int t = 0; t < 3; ++t) {
          const int colg = (g * 3 + t) * 16 + (lane & 15);
          const int scol = (colg < 48) ? colg : colg - 48;
          const v8h skip = *(const v8h*)&sH2t[scol][mrow];  // one b128 load
          v8h gpack;
#pragma unroll
          for (int r = 0; r < 8; ++r)
            gpack[r] = (_Float16)(fast_tanh(c[t][r]) + (float)skip[r]);
          *(v8h*)&sGt[colg][mrow] = gpack;   // one b128 store
        }
      }
    }

    // ---- T = env^T @ G: batch 6 WMMAs, then partial stores ----
    {
      const v16h Ae = load_a_tile(&sEnvT[0][0], ETS, 0, wave * 32, lane);
      v8f tc[6];
#pragma unroll
      for (int nt = 0; nt < 6; ++nt) {
        const v16h Bg = load_b_tile(&sGt[0][0], GTS, nt * 16, wave * 32, lane);
        v8f z = {};
        tc[nt] = wmma_f16(Ae, Bg, z);
      }
      if (lane < 16) {
#pragma unroll
        for (int nt = 0; nt < 6; ++nt)
#pragma unroll
          for (int r = 0; r < 4; ++r)
            sTpart[wave][r][nt * 16 + lane] = tc[nt][r];
      }
    }
    __syncthreads();

    // reduce partial T across waves, scale by 1/NNEI
    for (int o = tid; o < 4 * 96; o += 128) {
      const int cch = o / 96, m = o - cch * 96;
      sT[cch][m] = (sTpart[0][cch][m] + sTpart[1][cch][m] +
                    sTpart[2][cch][m] + sTpart[3][cch][m]) * (1.0f / 128.0f);
    }
    __syncthreads();

    // ---- D = T^T T[:, :8] -> 768 floats ----
    float* op = out + (size_t)site * (96 * AXIS);
    for (int o = tid; o < 96 * AXIS; o += 128) {
      const int m = o >> 3, k = o & 7;
      op[o] = sT[0][m] * sT[0][k] + sT[1][m] * sT[1][k] +
              sT[2][m] * sT[2][k] + sT[3][m] * sT[3][k];
    }
    __syncthreads();
  }
}

extern "C" void kernel_launch(void* const* d_in, const int* in_sizes, int n_in,
                              void* d_out, int out_size, void* d_ws, size_t ws_size,
                              hipStream_t stream) {
  const float* coord = (const float*)d_in[0];
  const float* davg  = (const float*)d_in[1];
  const float* dstd  = (const float*)d_in[2];
  const float* w1    = (const float*)d_in[3];
  const float* b1    = (const float*)d_in[4];
  const float* w2    = (const float*)d_in[5];
  const float* b2    = (const float*)d_in[6];
  const float* w3    = (const float*)d_in[7];
  const float* b3    = (const float*)d_in[8];
  const int*   nlist = (const int*)d_in[9];
  float* outp = (float*)d_out;

  dim3 grid(1024);
  dim3 block(128);
  hipLaunchKernelGGL(descrpt_se_a_kernel, grid, block, 0, stream,
                     coord, davg, dstd, w1, b1, w2, b2, w3, b3, nlist, outp);
}
